// Encoder_41257455845717
// MI455X (gfx1250) — matrix operand order repaired and checked by reading, not compiled, not run on hardware
//
#include <hip/hip_runtime.h>
#include <hip/hip_bf16.h>
#include <math.h>

#define B_  2
#define S_  2048
#define D_  256
#define H_  8
#define DK_ 32
#define DV_ 32
#define L_  3

typedef __attribute__((ext_vector_type(16))) __bf16    v16bf;
typedef __attribute__((ext_vector_type(8)))  float     v8f;
typedef __attribute__((ext_vector_type(8)))  unsigned  v8u;

// two f32 -> packed bf16x2 (truncate) in one v_perm_b32
__device__ __forceinline__ unsigned pack_bf16x2(float a, float b) {
    return __builtin_amdgcn_perm(__builtin_bit_cast(unsigned, b),
                                 __builtin_bit_cast(unsigned, a),
                                 0x07060302u);
}
__device__ __forceinline__ __bf16 f2bf(float f) {
    unsigned short h = (unsigned short)(__builtin_bit_cast(unsigned, f) >> 16);
    return __builtin_bit_cast(__bf16, h);
}
__device__ __forceinline__ float bfl(unsigned w) {           // low bf16 of word
    return __builtin_bit_cast(float, w << 16);
}
__device__ __forceinline__ float bfh(unsigned w) {           // high bf16 of word
    return __builtin_bit_cast(float, w & 0xFFFF0000u);
}
__device__ __forceinline__ v8u frag128(const __bf16* p0, const __bf16* p1) {
    const uint4 a = *(const uint4*)p0;                       // 8 bf16
    const uint4 b = *(const uint4*)p1;                       // 8 bf16
    return (v8u){a.x, a.y, a.z, a.w, b.x, b.y, b.z, b.w};
}

// ---------------------------------------------------------------------------
// GEMM C[M,N] = A[M,K] (bf16 row-major) * Bt[N,K] (bf16, pre-transposed).
// All fragment loads are contiguous global_load_b128.
// MODE 0: C bf16 row-major   MODE 1: C f32 row-major
// MODE 2: C bf16 per-head transposed vT[((b*H+h)*DV + c)*S + s]  (V projection)
// ---------------------------------------------------------------------------
template <int MODE>
__global__ __launch_bounds__(256) void gemm16_wmma_bf16(
    const __bf16* __restrict__ A, const __bf16* __restrict__ Bt,
    void* __restrict__ Cv, int M, int N, int K, int lda, int ldb, int ldc)
{
    const int wave   = threadIdx.x >> 5;
    const int lane   = threadIdx.x & 31;
    const int tilesN = N >> 4;
    const int tile   = blockIdx.x * (blockDim.x >> 5) + wave;
    if (tile >= (M >> 4) * tilesN) return;                   // wave-uniform
    const int tm = tile / tilesN;
    const int tn = tile - tm * tilesN;

    // wave32 fragment layouts used here (A/B k order is this code's contiguous choice, not the ISA table order):
    //  A 16x32: lanes 0-15 row=lane, K{0..15}; lanes 16-31 row=lane-16, K{16..31}
    //  B 32x16: lanes 0-15 col=lane K{0..15}; lanes 16-31 col=lane-16 K{16..31}
    //  C 16x16: VGPR r -> M = r + 8*(lane>=16), N = lane&15
    const int arow  = lane & 15;
    const int akoff = (lane >> 4) << 4;
    const int bn    = lane & 15;
    const int bkoff = (lane >> 4) << 4;

    const __bf16* Arow = A  + (long long)(tm * 16 + arow) * lda;
    const __bf16* Brow = Bt + (long long)(tn * 16 + bn)  * ldb;

    v8f acc = {};
    for (int kb = 0; kb < K; kb += 32) {
        const v8u au = frag128(Arow + kb + akoff, Arow + kb + akoff + 8);
        const v8u bu = frag128(Brow + kb + bkoff, Brow + kb + bkoff + 8);
        acc = __builtin_amdgcn_wmma_f32_16x16x32_bf16(
                  false, __builtin_bit_cast(v16bf, au),
                  false, __builtin_bit_cast(v16bf, bu),
                  (short)0, acc, false, false);
    }

    const int mo = (lane >> 4) << 3;
    const int n  = lane & 15;
#pragma unroll
    for (int r = 0; r < 8; ++r) {
        if constexpr (MODE == 0) {
            ((__bf16*)Cv)[(long long)(tm * 16 + r + mo) * ldc + (tn << 4) + n] = f2bf(acc[r]);
        } else if constexpr (MODE == 1) {
            ((float*)Cv)[(long long)(tm * 16 + r + mo) * ldc + (tn << 4) + n] = acc[r];
        } else {                                   // per-head transposed V
            const int mG = tm * 16 + r + mo;       // fused b*S + s
            const int jG = (tn << 4) + n;          // h*DV + c
            const int b  = mG >> 11, s = mG & (S_ - 1);
            const int h  = jG >> 5,  c = jG & 31;
            ((__bf16*)Cv)[((long long)((b * H_ + h) * DV_ + c) << 11) + s] = f2bf(acc[r]);
        }
    }
}

// ---------------------------------------------------------------------------
// scores = (Q_h @ K_h^T)/sqrt(DK) per (b,h); K=32 -> single WMMA; bf16 output.
// K-buffer is row-major [B*S, D], which IS the N x K layout scores need.
// ---------------------------------------------------------------------------
__global__ __launch_bounds__(256) void scores_wmma(
    const __bf16* __restrict__ qb, const __bf16* __restrict__ kb,
    __bf16* __restrict__ sraw, float scale)
{
    const int wave = threadIdx.x >> 5;
    const int lane = threadIdx.x & 31;
    const int tile = blockIdx.x * (blockDim.x >> 5) + wave;
    const int tm   = tile >> 7;                       // S/16 = 128
    const int tn   = tile & 127;
    const int bh   = blockIdx.z * H_ + blockIdx.y;

    const int arow  = lane & 15;
    const int akoff = (lane >> 4) << 4;
    const int bn    = lane & 15;
    const int bkoff = (lane >> 4) << 4;

    const __bf16* Q  = qb + ((long long)blockIdx.z * S_ + tm * 16 + arow) * D_ + blockIdx.y * DK_;
    const __bf16* Kp = kb + ((long long)blockIdx.z * S_ + tn * 16 + bn)  * D_ + blockIdx.y * DK_;

    const v8u au = frag128(Q + akoff,  Q + akoff + 8);
    const v8u bu = frag128(Kp + bkoff, Kp + bkoff + 8);
    v8f acc = {};
    acc = __builtin_amdgcn_wmma_f32_16x16x32_bf16(
              false, __builtin_bit_cast(v16bf, au),
              false, __builtin_bit_cast(v16bf, bu),
              (short)0, acc, false, false);

    __bf16* Cp = sraw + (long long)bh * S_ * S_ + (long long)(tm << 4) * S_ + (tn << 4);
    const int mo = (lane >> 4) << 3;
    const int n  = lane & 15;
#pragma unroll
    for (int r = 0; r < 8; ++r)
        Cp[(long long)(r + mo) * S_ + n] = f2bf(acc[r] * scale);
}

// ---------------------------------------------------------------------------
// One block per row, row held in registers (8 elems/thread via b128):
// softmax; write normalized f32 to d_out (only touch) + bf16 in place for PV.
// ---------------------------------------------------------------------------
__global__ __launch_bounds__(256) void softmax_rows(
    __bf16* __restrict__ sraw, float* __restrict__ attnF)
{
    __shared__ float sred[256];
    const long long row = blockIdx.x;
    __bf16* pr = sraw  + row * S_ + threadIdx.x * 8;
    float*  pf = attnF + row * S_ + threadIdx.x * 8;
    const int tid = threadIdx.x;

    const uint4 w = *(const uint4*)pr;
    float e[8] = { bfl(w.x), bfh(w.x), bfl(w.y), bfh(w.y),
                   bfl(w.z), bfh(w.z), bfl(w.w), bfh(w.w) };

    float m = e[0];
#pragma unroll
    for (int t = 1; t < 8; ++t) m = fmaxf(m, e[t]);
    sred[tid] = m;
    __syncthreads();
    for (int off = 128; off > 0; off >>= 1) {
        if (tid < off) sred[tid] = fmaxf(sred[tid], sred[tid + off]);
        __syncthreads();
    }
    m = sred[0];
    __syncthreads();

    float s = 0.f;
#pragma unroll
    for (int t = 0; t < 8; ++t) { e[t] = __expf(e[t] - m); s += e[t]; }
    sred[tid] = s;
    __syncthreads();
    for (int off = 128; off > 0; off >>= 1) {
        if (tid < off) sred[tid] += sred[tid + off];
        __syncthreads();
    }
    const float inv = 1.0f / sred[0];

#pragma unroll
    for (int t = 0; t < 8; ++t) e[t] *= inv;
    ((float4*)pf)[0] = make_float4(e[0], e[1], e[2], e[3]);
    ((float4*)pf)[1] = make_float4(e[4], e[5], e[6], e[7]);
    uint4 o;
    o.x = pack_bf16x2(e[0], e[1]); o.y = pack_bf16x2(e[2], e[3]);
    o.z = pack_bf16x2(e[4], e[5]); o.w = pack_bf16x2(e[6], e[7]);
    *(uint4*)pr = o;
}

// ---------------------------------------------------------------------------
// ctx = attn_norm @ V_h per (b,h): wave owns 16x32 slab, 2 WMMAs per k-step.
// A = normalized bf16 attn (sraw), B = per-head transposed V -> all b128 loads.
// ---------------------------------------------------------------------------
__global__ __launch_bounds__(256) void ctx_wmma(
    const __bf16* __restrict__ sraw, const __bf16* __restrict__ vt,
    __bf16* __restrict__ ctxb)
{
    const int wave = threadIdx.x >> 5;
    const int lane = threadIdx.x & 31;
    const int tm   = blockIdx.x * (blockDim.x >> 5) + wave;   // 0..127
    const int bh   = blockIdx.z * H_ + blockIdx.y;

    const int arow  = lane & 15;
    const int akoff = (lane >> 4) << 4;
    const int bn    = lane & 15;
    const int bkoff = (lane >> 4) << 4;

    const __bf16* Arow = sraw + (long long)bh * S_ * S_ + (long long)(tm * 16 + arow) * S_;
    const __bf16* V0   = vt + ((long long)bh * DV_ + bn) * S_;        // cols 0..15
    const __bf16* V1   = V0 + (long long)16 * S_;                      // cols 16..31

    v8f acc0 = {}, acc1 = {};
    for (int kb = 0; kb < S_; kb += 32) {
        const v8u au  = frag128(Arow + kb + akoff, Arow + kb + akoff + 8);
        const v8u bu0 = frag128(V0 + kb + bkoff,   V0 + kb + bkoff + 8);
        const v8u bu1 = frag128(V1 + kb + bkoff,   V1 + kb + bkoff + 8);
        const v16bf a = __builtin_bit_cast(v16bf, au);
        acc0 = __builtin_amdgcn_wmma_f32_16x16x32_bf16(
                   false, a, false, __builtin_bit_cast(v16bf, bu0), (short)0, acc0, false, false);
        acc1 = __builtin_amdgcn_wmma_f32_16x16x32_bf16(
                   false, a, false, __builtin_bit_cast(v16bf, bu1), (short)0, acc1, false, false);
    }

    const int mo = (lane >> 4) << 3;
    const int n  = lane & 15;
    __bf16* Cp = ctxb + ((long long)blockIdx.z * S_ + (tm << 4)) * D_ + blockIdx.y * DV_;
#pragma unroll
    for (int r = 0; r < 8; ++r) {
        Cp[(long long)(r + mo) * D_ + n]      = f2bf(acc0[r]);
        Cp[(long long)(r + mo) * D_ + 16 + n] = f2bf(acc1[r]);
    }
}

// ---------------------------------------------------------------------------
__global__ __launch_bounds__(256) void add_pos_enc(const float* __restrict__ in,
                                                   __bf16* __restrict__ xb)
{
    const long long idx = (long long)blockIdx.x * blockDim.x + threadIdx.x;
    if (idx >= (long long)B_ * S_ * D_) return;
    const int d  = (int)(idx % D_);
    const int s  = (int)((idx / D_) % S_);
    const int i2 = d & ~1;
    const float dv  = __expf((float)i2 * (-9.210340371976184f / (float)D_));
    const float ang = (float)s * dv;
    const float pe  = (d & 1) ? __cosf(ang) : __sinf(ang);
    xb[idx] = f2bf(in[idx] + pe);
}

// W[K,N] f32 -> Wt[N,K] bf16 (one-time per layer; reused 256x by the GEMMs)
__global__ __launch_bounds__(256) void transpose_w_bf16(
    const float* __restrict__ W, __bf16* __restrict__ Wt, int K, int N)
{
    const int i = blockIdx.x * blockDim.x + threadIdx.x;   // i = n*K + k
    if (i >= K * N) return;
    const int n = i / K, k = i - n * K;
    Wt[i] = f2bf(W[(long long)k * N + n]);
}

// ---------------------------------------------------------------------------
extern "C" void kernel_launch(void* const* d_in, const int* in_sizes, int n_in,
                              void* d_out, int out_size, void* d_ws, size_t ws_size,
                              hipStream_t stream)
{
    const float* seq = (const float*)d_in[0];
    const float* Wq  = (const float*)d_in[1];
    const float* Wk  = (const float*)d_in[2];
    const float* Wv  = (const float*)d_in[3];
    const float* Wfc = (const float*)d_in[4];

    float* out   = (float*)d_out;
    float* attns = out + (size_t)B_ * S_ * D_;           // [L,B,H,S,S] f32

    const size_t nBSD = (size_t)B_ * S_ * D_;            // 1,048,576
    const size_t nWW  = (size_t)D_ * D_;                 // 65,536
    __bf16* xb   = (__bf16*)d_ws;
    __bf16* qb   = xb   + nBSD;
    __bf16* kbuf = qb   + nBSD;
    __bf16* vt   = kbuf + nBSD;                          // [B,H,DV,S]
    __bf16* ctxb = vt   + nBSD;
    __bf16* wqt  = ctxb + nBSD;
    __bf16* wkt  = wqt  + nWW;
    __bf16* wvt  = wkt  + nWW;
    __bf16* wft  = wvt  + nWW;
    __bf16* sraw = wft  + nWW;                           // [B,H,S,S] bf16 scores

    const float inv_sqrt_dk = 0.17677669529663687f;
    const int   MB   = B_ * S_;                          // 4096
    const int   nRow = B_ * H_ * S_;                     // 32768

    add_pos_enc<<<(B_ * S_ * D_ + 255) / 256, 256, 0, stream>>>(seq, xb);

    const dim3 gProj((MB / 16) * (D_ / 16) / 8, 1, 1);   // 512
    const dim3 gScor((S_ / 16) * (S_ / 16) / 8, H_, B_); // 2048 x 8 x 2
    const dim3 gCtx((S_ / 16) / 8, H_, B_);              // 16 x 8 x 2
    const int  gW = (int)((nWW + 255) / 256);

    for (int l = 0; l < L_; ++l) {
        transpose_w_bf16<<<gW, 256, 0, stream>>>(Wq  + (size_t)l * nWW, wqt, D_, D_);
        transpose_w_bf16<<<gW, 256, 0, stream>>>(Wk  + (size_t)l * nWW, wkt, D_, D_);
        transpose_w_bf16<<<gW, 256, 0, stream>>>(Wv  + (size_t)l * nWW, wvt, D_, D_);
        transpose_w_bf16<<<gW, 256, 0, stream>>>(Wfc + (size_t)l * nWW, wft, D_, D_);

        gemm16_wmma_bf16<0><<<gProj, 256, 0, stream>>>(xb, wqt, qb,   MB, D_, D_, D_, D_, D_);
        gemm16_wmma_bf16<0><<<gProj, 256, 0, stream>>>(xb, wkt, kbuf, MB, D_, D_, D_, D_, D_);
        gemm16_wmma_bf16<2><<<gProj, 256, 0, stream>>>(xb, wvt, vt,   MB, D_, D_, D_, D_, D_);

        float* attn_l = attns + (size_t)l * B_ * H_ * S_ * S_;
        scores_wmma <<<gScor, 256, 0, stream>>>(qb, kbuf, sraw, inv_sqrt_dk);
        softmax_rows<<<nRow, 256, 0, stream>>>(sraw, attn_l);
        ctx_wmma    <<<gCtx, 256, 0, stream>>>(sraw, vt, ctxb);

        if (l == L_ - 1)
            gemm16_wmma_bf16<1><<<gProj, 256, 0, stream>>>(ctxb, wft, out, MB, D_, D_, D_, D_, D_);
        else
            gemm16_wmma_bf16<0><<<gProj, 256, 0, stream>>>(ctxb, wft, xb,  MB, D_, D_, D_, D_, D_);
    }
}
